// OlmoLMHeadCE_42674795053758
// MI455X (gfx1250) — matrix operand order repaired and checked by reading, not compiled, not run on hardware
//
#include <hip/hip_runtime.h>
#include <cmath>

#define IGNORE_INDEX (-100)

typedef __attribute__((ext_vector_type(16))) __bf16 v16bf;
typedef __attribute__((ext_vector_type(8)))  float  v8f;
typedef unsigned int u32x4 __attribute__((ext_vector_type(4)));
typedef int          i32x8 __attribute__((ext_vector_type(8)));
typedef int          i32x4 __attribute__((ext_vector_type(4)));

constexpr int kN = 4096;
constexpr int kH = 1024;
constexpr int kV = 32000;
constexpr int ROWS      = 32;           // rows per block (two 16-row M tiles)
constexpr int WAVES     = 8;            // 256 threads, wave32
constexpr int THREADS   = WAVES * 32;
constexpr int COLS_ITER = WAVES * 16;   // 128 columns per block iteration
constexpr int CSPLIT    = 8;            // V-dimension splits
constexpr int SPLITC    = 4096;         // cols per split (last split: 3328 = 26*128)
constexpr int KC        = 128;          // K-chunk staged per TDM transfer
constexpr int NCHUNK    = kH / KC;      // 8
constexpr int WROW      = 130;          // padded LDS row stride (floats): bank-conflict free

// LDS layout (bytes)
constexpr unsigned XHI_OFF  = 0;
constexpr unsigned XLO_OFF  = XHI_OFF + ROWS * kH * 2;          // 65536
constexpr unsigned WBUF_OFF = XLO_OFF + ROWS * kH * 2;          // 131072
constexpr unsigned WBYTES   = KC * WROW * 4;                    // 66560 per buffer
constexpr unsigned COMB_OFF = WBUF_OFF + 2 * WBYTES;            // 264192
constexpr unsigned LDS_TOTAL = COMB_OFF + WAVES * 32 * 3 * 4;   // 267264 (< 320 KB)

__device__ __forceinline__ float ninf() { return -__builtin_inff(); }

// Load a 16-half A fragment (two 16-byte LDS chunks: K..K+7 and K+16..K+23)
__device__ __forceinline__ v16bf load_afrag(const __bf16* base) {
  union { v16bf v; uint4 q[2]; } u;
  u.q[0] = *(const uint4*)(base);
  u.q[1] = *(const uint4*)(base + 16);
  return u.v;
}

// Issue one async TDM transfer: 2D fp32 tile [KC rows x 128 cols] of W -> LDS,
// with LDS padding 2 dwords per 128 dwords (row stride 130 floats).
__device__ __forceinline__ void tdm_issue_w(const float* gaddr, unsigned lds_off) {
  const unsigned long long ga = (unsigned long long)(size_t)gaddr;
  union { unsigned u[4]; u32x4 v; } g0 = {};
  g0.u[0] = 1u;                                   // count=1 valid descriptor, user mode
  g0.u[1] = lds_off;                              // lds_addr (bytes)
  g0.u[2] = (unsigned)(ga & 0xFFFFFFFFu);         // global_addr[31:0]
  g0.u[3] = (unsigned)((ga >> 32) & 0x01FFFFFFu)  // global_addr[56:32]
          | (2u << 30);                           // type = 2 ("image")
  union { unsigned u[8]; i32x8 v; } g1 = {};
  g1.u[0] = (2u << 16)                            // data_size = 4 bytes
          | (1u << 20)                            // pad_enable
          | (6u << 22)                            // pad_interval: every 128 dwords
          | (1u << 25);                           // pad_amount: 2 dwords
  g1.u[1] = ((unsigned)kV & 0xFFFFu) << 16;       // tensor_dim0[15:0]
  g1.u[2] = ((unsigned)kV >> 16)                  // tensor_dim0[31:16]
          | (((unsigned)kH & 0xFFFFu) << 16);     // tensor_dim1[15:0]
  g1.u[3] = ((unsigned)kH >> 16)
          | ((unsigned)COLS_ITER << 16);          // tile_dim0 = 128
  g1.u[4] = (unsigned)KC;                         // tile_dim1 = 128, tile_dim2 = 0
  g1.u[5] = (unsigned)kV;                         // tensor_dim0_stride[31:0] = V
  g1.u[6] = 0;                                    // stride0 hi / stride1 lo (unused for 2D)
  g1.u[7] = 0;
  i32x4 z4; z4[0] = 0; z4[1] = 0; z4[2] = 0; z4[3] = 0;
  i32x8 z8; 
  #pragma unroll
  for (int i = 0; i < 8; ++i) z8[i] = 0;
  // 6-arg form (clang-23 / therock HIP headers): (g0, g1, g2, g3, g4, cpol)
  __builtin_amdgcn_tensor_load_to_lds(g0.v, g1.v, z4, z4, z8, 0);
}

__global__ void __launch_bounds__(THREADS)
lmhead_ce_partial(const float* __restrict__ x, const float* __restrict__ W,
                  const long long* __restrict__ target,
                  float* __restrict__ pm, float* __restrict__ ps,
                  float* __restrict__ pt)
{
  extern __shared__ char smem[];
  __bf16* xhi  = (__bf16*)(smem + XHI_OFF);       // [ROWS][kH] bf16 hi
  __bf16* xlo  = (__bf16*)(smem + XLO_OFF);       // [ROWS][kH] bf16 lo
  float*  comb = (float*)(smem + COMB_OFF);       // [WAVES][32][3]
  const unsigned wb_base = (unsigned)(size_t)(smem + WBUF_OFF); // LDS byte offset of W ring

  const int split   = blockIdx.x;
  const int rowbase = blockIdx.y * ROWS;
  const int tid  = threadIdx.x;
  const int wave = tid >> 5;
  const int lane = tid & 31;
  const int laneHi = lane >> 4;
  const int ln     = lane & 15;

  const int colstart = split * SPLITC;
  const int colcount = (split == CSPLIT - 1) ? (kV - (CSPLIT - 1) * SPLITC) : SPLITC;
  const int ntiles   = colcount / COLS_ITER;      // exact: 32 or 26

  // Kick the first TDM transfer immediately so the DMA overlaps x staging.
  if (wave == 0) tdm_issue_w(&W[(size_t)0 * kV + colstart], wb_base);

  // ---- Stage x tile into LDS as bf16 hi/lo (fp32 -> bf16x2 split) ----
  for (int e = tid; e < ROWS * kH; e += THREADS) {
    const int r = e >> 10;            // kH == 1024
    const int k = e & (kH - 1);
    const float f = x[(size_t)(rowbase + r) * kH + k];
    const __bf16 h = (__bf16)f;
    xhi[e] = h;
    xlo[e] = (__bf16)(f - (float)h);
  }
  __syncthreads();

  // lane r owns the running online-softmax state of row (rowbase + r)
  const int tgt = (int)target[rowbase + lane];
  float st_m = ninf(), st_s = 0.f, st_t = ninf();

  for (int it = 0; it < ntiles; ++it) {
    const int cbBlock = colstart + it * COLS_ITER;
    const int mycol   = cbBlock + wave * 16 + ln;

    v8f acc0, acc1;
    #pragma unroll
    for (int i = 0; i < 8; ++i) { acc0[i] = 0.f; acc1[i] = 0.f; }

    for (int c = 0; c < NCHUNK; ++c) {
      if (wave == 0) __builtin_amdgcn_s_wait_tensorcnt(0);  // chunk c landed in LDS
      __syncthreads();                                      // publish buffer to all waves
      if (wave == 0) {
        // Issue the next chunk (same buffer parity pattern); it overlaps compute below.
        if (c + 1 < NCHUNK)
          tdm_issue_w(&W[(size_t)((c + 1) * KC) * kV + cbBlock],
                      wb_base + (unsigned)(((c + 1) & 1) * WBYTES));
        else if (it + 1 < ntiles)
          tdm_issue_w(&W[(size_t)0 * kV + cbBlock + COLS_ITER], wb_base); // next tile chunk 0
      }

      const float* wb = (const float*)(smem + WBUF_OFF + (size_t)(c & 1) * WBYTES);
      const int kglob = c * KC;

      for (int kk = 0; kk < KC; kk += 32) {
        // A fragments (16x32 bf16): lanes 16-31 take the K+8 / K+24 halves
        const int kbA = kglob + kk + (laneHi << 3);
        const v16bf a0h = load_afrag(&xhi[(size_t)ln * kH + kbA]);
        const v16bf a0l = load_afrag(&xlo[(size_t)ln * kH + kbA]);
        const v16bf a1h = load_afrag(&xhi[(size_t)(ln + 16) * kH + kbA]);
        const v16bf a1l = load_afrag(&xlo[(size_t)(ln + 16) * kH + kbA]);

        // B fragment (32x16 bf16) from the TDM-staged LDS tile (stride 130: no bank conflicts)
        const int bbase = (kk + (laneHi << 3)) * WROW + wave * 16 + ln;
        v16bf bh, bl;
        #pragma unroll
        for (int i = 0; i < 16; ++i) {
          const float w = wb[bbase + (i + ((i >> 3) << 3)) * WROW];
          const __bf16 h = (__bf16)w;
          bh[i] = h;
          bl[i] = (__bf16)(w - (float)h);
        }

        // bf16x3 fp32-emulated GEMM: hi*hi + hi*lo + lo*hi, fp32 accumulate
        acc0 = __builtin_amdgcn_wmma_f32_16x16x32_bf16(false, a0h, false, bh, (short)0, acc0, false, false);
        acc0 = __builtin_amdgcn_wmma_f32_16x16x32_bf16(false, a0h, false, bl, (short)0, acc0, false, false);
        acc0 = __builtin_amdgcn_wmma_f32_16x16x32_bf16(false, a0l, false, bh, (short)0, acc0, false, false);
        acc1 = __builtin_amdgcn_wmma_f32_16x16x32_bf16(false, a1h, false, bh, (short)0, acc1, false, false);
        acc1 = __builtin_amdgcn_wmma_f32_16x16x32_bf16(false, a1h, false, bl, (short)0, acc1, false, false);
        acc1 = __builtin_amdgcn_wmma_f32_16x16x32_bf16(false, a1l, false, bh, (short)0, acc1, false, false);
      }
    }

    // ---- Online logsumexp update of 32x16 logits ----
    // C/D layout: VGPR j, lanes 0-15 -> row T*16+j, lanes 16-31 -> row T*16+8+j, col = lane&15
    #pragma unroll
    for (int T = 0; T < 2; ++T) {
      const v8f acc = T ? acc1 : acc0;
      #pragma unroll
      for (int j = 0; j < 8; ++j) {
        const float v = acc[j];
        float rmax = v;
        #pragma unroll
        for (int d = 1; d < 16; d <<= 1) rmax = fmaxf(rmax, __shfl_xor(rmax, d, 32));
        float rsum = __expf(v - rmax);
        #pragma unroll
        for (int d = 1; d < 16; d <<= 1) rsum += __shfl_xor(rsum, d, 32);

        const int myrow = T * 16 + j + (laneHi << 3);
        const int trow  = __shfl(tgt, myrow, 32);          // target col for my row
        float tv = (mycol == trow) ? v : ninf();
        #pragma unroll
        for (int d = 1; d < 16; d <<= 1) tv = fmaxf(tv, __shfl_xor(tv, d, 32));

        // deliver to owner lanes r0 (value lives in lanes 0-15) and r1 (lanes 16-31)
        const int r0 = T * 16 + j, r1 = r0 + 8;
        const int src = (lane == r1) ? 16 : 0;
        const float gm = __shfl(rmax, src, 32);
        const float gs = __shfl(rsum, src, 32);
        const float gt = __shfl(tv,   src, 32);
        if (lane == r0 || lane == r1) {
          const float nm = fmaxf(st_m, gm);
          st_s = st_s * __expf(st_m - nm) + gs * __expf(gm - nm);
          st_m = nm;
          st_t = fmaxf(st_t, gt);
        }
      }
    }
  }

  // ---- Combine the 8 waves' per-row partial states, write split partials ----
  comb[(wave * 32 + lane) * 3 + 0] = st_m;
  comb[(wave * 32 + lane) * 3 + 1] = st_s;
  comb[(wave * 32 + lane) * 3 + 2] = st_t;
  __syncthreads();
  if (wave == 0) {
    float M = ninf(), S = 0.f, T = ninf();
    for (int w = 0; w < WAVES; ++w) {
      const float m = comb[(w * 32 + lane) * 3 + 0];
      const float s = comb[(w * 32 + lane) * 3 + 1];
      const float t = comb[(w * 32 + lane) * 3 + 2];
      const float nm = fmaxf(M, m);
      S = S * __expf(M - nm) + s * __expf(m - nm);
      M = nm;
      T = fmaxf(T, t);
    }
    const size_t idx = (size_t)split * kN + rowbase + lane;
    pm[idx] = M; ps[idx] = S; pt[idx] = T;
  }
}

__global__ void __launch_bounds__(256)
lmhead_ce_reduce(const float* __restrict__ pm, const float* __restrict__ ps,
                 const float* __restrict__ pt, const long long* __restrict__ target,
                 float* __restrict__ out)
{
  __shared__ float red[256];
  const int tid = threadIdx.x;
  float local = 0.f;
  for (int row = tid; row < kN; row += 256) {
    if (target[row] == IGNORE_INDEX) continue;
    float M = ninf(), S = 0.f, T = ninf();
    for (int s = 0; s < CSPLIT; ++s) {
      const size_t idx = (size_t)s * kN + row;
      const float m = pm[idx], sv = ps[idx], t = pt[idx];
      const float nm = fmaxf(M, m);
      S = S * __expf(M - nm) + sv * __expf(m - nm);
      M = nm;
      T = fmaxf(T, t);
    }
    local += (M + __logf(S)) - T;   // lse - target_logit
  }
  red[tid] = local;
  __syncthreads();
  for (int off = 128; off > 0; off >>= 1) {
    if (tid < off) red[tid] += red[tid + off];
    __syncthreads();
  }
  if (tid == 0) out[0] = red[0];
}

extern "C" void kernel_launch(void* const* d_in, const int* in_sizes, int n_in,
                              void* d_out, int out_size, void* d_ws, size_t ws_size,
                              hipStream_t stream) {
  const float*      x      = (const float*)d_in[0];
  const float*      W      = (const float*)d_in[1];
  const long long*  target = (const long long*)d_in[2];

  float* pm = (float*)d_ws;                    // [CSPLIT][kN] running max
  float* ps = pm + (size_t)CSPLIT * kN;        // [CSPLIT][kN] running sum
  float* pt = ps + (size_t)CSPLIT * kN;        // [CSPLIT][kN] target logit (or -inf)

  const dim3 grid(CSPLIT, kN / ROWS);          // 8 x 128 workgroups
  lmhead_ce_partial<<<grid, THREADS, LDS_TOTAL, stream>>>(x, W, target, pm, ps, pt);
  lmhead_ce_reduce<<<1, 256, 0, stream>>>(pm, ps, pt, target, (float*)d_out);
}
